// Seq2Seq_81673098101020
// MI455X (gfx1250) — matrix operand order repaired and checked by reading, not compiled, not run on hardware
//
#include <hip/hip_runtime.h>
#include <hip/hip_bf16.h>

typedef __attribute__((ext_vector_type(16))) _Float16 v16h;
typedef __attribute__((ext_vector_type(8)))  float    v8f;

// ---------------------------------------------------------------------------
// WMMA helpers (CDNA5: D = A(16x32 f16) * B(32x16 f16) + C(16x16 f32))
// ---------------------------------------------------------------------------
__device__ __forceinline__ v8f wmma16(v16h a, v16h b, v8f c) {
    return __builtin_amdgcn_wmma_f32_16x16x32_f16(
        /*neg_a=*/false, a, /*neg_b=*/false, b,
        /*c_mod=*/(short)0, c, /*reuse_a=*/false, /*reuse_b=*/false);
}

__device__ __forceinline__ v8f vinit(float x) {
    v8f r;
#pragma unroll
    for (int v = 0; v < 8; ++v) r[v] = x;
    return r;
}

// A fragment: 16x32 f16 tile, row-major source with leading dim ld.
// Contiguous k order (same as B): lane m=L&15, hi=L>>4; VGPR v holds K pair kb,kb+1 with
// kb = 2v + 16*hi.
__device__ __forceinline__ v16h load_a_frag(const _Float16* A, int ld,
                                            int m, int hi, int k0) {
    v16h f;
    const _Float16* r = A + (size_t)m * ld + k0;
#pragma unroll
    for (int v = 0; v < 8; ++v) {
        const int kb = 2 * v + hi * 16;
        f[2 * v]     = r[kb];
        f[2 * v + 1] = r[kb + 1];
    }
    return f;
}

// B fragment: 32x16 f16 tile where B[k][n] = W[n][k] (W row-major, ld = K).
// Contiguous k order (same as A): lane n=L&15, hi=L>>4; halves i=0..15 hold K = k0+16*hi+i.
// => one contiguous 32-byte read.
__device__ __forceinline__ v16h load_b_frag(const _Float16* W, int ld,
                                            int ncol, int hi, int k0) {
    return *(const v16h*)(W + (size_t)ncol * ld + k0 + hi * 16);
}

// Accumulate the three gate tiles (cols c0, 128+c0, 256+c0 of a 384-row
// weight matrix) for one A operand: o{0,1,2} += A * W^T.
__device__ __forceinline__ void mm3(const _Float16* A, int lda,
                                    const _Float16* W, int ldw, int Kin,
                                    int c0, int nn, int hi,
                                    v8f& o0, v8f& o1, v8f& o2) {
    for (int kk = 0; kk < Kin; kk += 32) {
        v16h a = load_a_frag(A, lda, nn, hi, kk);
        o0 = wmma16(a, load_b_frag(W, ldw, c0,       hi, kk), o0);
        o1 = wmma16(a, load_b_frag(W, ldw, 128 + c0, hi, kk), o1);
        o2 = wmma16(a, load_b_frag(W, ldw, 256 + c0, hi, kk), o2);
    }
}

__device__ __forceinline__ float sigm(float x) { return 1.f / (1.f + __expf(-x)); }
__device__ __forceinline__ float tanh_f(float x) {
    float e = __expf(-2.f * x);
    return (1.f - e) / (1.f + e);
}

// ---------------------------------------------------------------------------
// f32 -> f16 weight conversion
// ---------------------------------------------------------------------------
__global__ void cvt_kernel(const float* __restrict__ in,
                           _Float16* __restrict__ out, int n) {
    int i = blockIdx.x * blockDim.x + threadIdx.x;
    if (i < n) out[i] = (_Float16)in[i];
}

// ---------------------------------------------------------------------------
// Encoder bidirectional GRU layer. grid = 64 blocks:
//   dir = blockIdx.x>>5 (0=fwd,1=bwd), batch tile = (blockIdx.x&31)*16.
// One barrier per timestep: h and xt are double-buffered (writes always go
// to the other buffer), and the next step's input tile is prefetched into
// registers during the WMMA/gate phase so L2 latency overlaps compute.
// ---------------------------------------------------------------------------
__global__ __launch_bounds__(256) void enc_layer_kernel(
    int layer, int Kin,
    const float*    __restrict__ x,      // (512,512,64) f32  (layer 0 input)
    _Float16*       __restrict__ out0,   // (512,512,256) f16 ws (l0 out / l1 in)
    const _Float16* __restrict__ Wih_h,  // (2,384,Kin) f16
    const _Float16* __restrict__ Whh_h,  // (2,2,384,128) f16 (indexed by layer)
    const float*    __restrict__ bih,    // (2,2,384) f32
    const float*    __restrict__ bhh,    // (2,2,384) f32
    _Float16*       __restrict__ hn)     // (4,512,128) f16
{
    extern __shared__ __align__(64) char smem[];
    _Float16* whh_l = (_Float16*)smem;          // 384*128
    _Float16* wih_l = whh_l + 384 * 128;        // 384*Kin
    _Float16* xt0   = wih_l + 384 * Kin;        // 16*Kin   (ping)
    _Float16* xt1   = xt0 + 16 * Kin;           // 16*Kin   (pong)
    _Float16* h0    = xt1 + 16 * Kin;           // 16*128   (ping)
    _Float16* h1    = h0 + 16 * 128;            // 16*128   (pong)

    const int tid  = threadIdx.x;
    const int wv   = tid >> 5;
    const int ln   = tid & 31;
    const int nn   = ln & 15;
    const int hi   = ln >> 4;
    const int dir  = blockIdx.x >> 5;
    const int row0 = (blockIdx.x & 31) * 16;
    const int c0   = wv * 16 + nn;

    // Stage weights into LDS (f16, row-major [384][K]).
    const _Float16* whh_g = Whh_h + (size_t)(layer * 2 + dir) * 384 * 128;
    for (int i = tid; i < 384 * 128; i += 256) whh_l[i] = whh_g[i];
    const _Float16* wih_g = Wih_h + (size_t)dir * 384 * Kin;
    for (int i = tid; i < 384 * Kin; i += 256) wih_l[i] = wih_g[i];
    for (int i = tid; i < 16 * 128; i += 256) h0[i] = (_Float16)0.f;

    const float* bih_p = bih + (size_t)(layer * 2 + dir) * 384;
    const float* bhh_p = bhh + (size_t)(layer * 2 + dir) * 384;
    const float b_ir = bih_p[c0], b_iz = bih_p[128 + c0], b_in = bih_p[256 + c0];
    const float b_hr = bhh_p[c0], b_hz = bhh_p[128 + c0], b_hn = bhh_p[256 + c0];

    // Stage step-0 input tile into xt0.
    {
        const int t0 = dir ? 511 : 0;
        if (layer == 0) {
            const float* src = x + ((size_t)t0 * 512 + row0) * 64;
            for (int i = tid; i < 16 * 64; i += 256) xt0[i] = (_Float16)src[i];
        } else {
            const unsigned* src = (const unsigned*)(out0 + ((size_t)t0 * 512 + row0) * 256);
            unsigned* dst = (unsigned*)xt0;
            for (int i = tid; i < 16 * 256 / 2; i += 256) dst[i] = src[i];
        }
    }
    __syncthreads();

    auto step_fn = [&](int s, _Float16* xcur, _Float16* xnxt,
                       _Float16* hcur, _Float16* hnxt) {
        const int tt = dir ? (511 - s) : s;

        // Prefetch next step's input tile into registers (overlaps compute).
        const int sn = (s < 511) ? (s + 1) : 511;
        const int tn = dir ? (511 - sn) : sn;
        float    pf_f[4];
        unsigned pf_u[8];
        if (layer == 0) {
            const float* src = x + ((size_t)tn * 512 + row0) * 64;
#pragma unroll
            for (int j = 0; j < 4; ++j) pf_f[j] = src[tid + j * 256];
        } else {
            const unsigned* src = (const unsigned*)(out0 + ((size_t)tn * 512 + row0) * 256);
#pragma unroll
            for (int j = 0; j < 8; ++j) pf_u[j] = src[tid + j * 256];
        }

        v8f a_r  = vinit(b_ir + b_hr);   // ir + hr
        v8f a_z  = vinit(b_iz + b_hz);   // iz + hz
        v8f a_in = vinit(b_in);          // i_n  (n = tanh(i_n + r*h_n))
        v8f a_hn = vinit(b_hn);          // h_n

        mm3(xcur, Kin, wih_l, Kin, Kin, c0, nn, hi, a_r, a_z, a_in);  // gi
        mm3(hcur, 128, whh_l, 128, 128, c0, nn, hi, a_r, a_z, a_hn);  // gh

        // Gate math; writes go to the *other* buffers -> no WAR hazard.
#pragma unroll
        for (int v = 0; v < 8; ++v) {
            const int m = v + 8 * hi;
            float r  = sigm(a_r[v]);
            float z  = sigm(a_z[v]);
            float ng = tanh_f(a_in[v] + r * a_hn[v]);
            float ho = (float)hcur[m * 128 + c0];
            _Float16 hv = (_Float16)((1.f - z) * ng + z * ho);
            hnxt[m * 128 + c0] = hv;
            if (layer == 0)
                out0[((size_t)tt * 512 + row0 + m) * 256 + dir * 128 + c0] = hv;
        }

        // Commit prefetched input tile for step s+1.
        if (layer == 0) {
#pragma unroll
            for (int j = 0; j < 4; ++j) xnxt[tid + j * 256] = (_Float16)pf_f[j];
        } else {
            unsigned* dst = (unsigned*)xnxt;
#pragma unroll
            for (int j = 0; j < 8; ++j) dst[tid + j * 256] = pf_u[j];
        }
        __syncthreads();   // one barrier per step
    };

    for (int s = 0; s < 512; s += 2) {
        step_fn(s,     xt0, xt1, h0, h1);
        step_fn(s + 1, xt1, xt0, h1, h0);
    }

    // Final hidden state lives in buffer 0 (after 512 steps).
    _Float16* dst = hn + ((size_t)(layer * 2 + dir) * 512 + row0) * 128;
    for (int i = tid; i < 16 * 128; i += 256) dst[i] = h0[i];
}

// ---------------------------------------------------------------------------
// Decoder: 24 sequential steps, 4 GRU cells per step + FC.
// grid = 32 blocks, each owns a 16-row batch tile; all state in LDS.
// Weights read as B-fragments straight from L2-resident f16 workspace.
// ---------------------------------------------------------------------------
__global__ __launch_bounds__(256) void dec_kernel(
    const float*    __restrict__ x,      // (512,512,64) f32 (for x0 = x[-1,:,0])
    const _Float16* __restrict__ hn,     // (4,512,128) f16
    const float*    __restrict__ dWih0,  // (2,384,1) f32
    const _Float16* __restrict__ dWih1,  // (2,384,256) f16
    const _Float16* __restrict__ dWhh,   // (2,2,384,128) f16
    const float*    __restrict__ dbih,   // (2,2,384) f32
    const float*    __restrict__ dbhh,   // (2,2,384) f32
    const float*    __restrict__ fcw,    // (256) f32
    const float*    __restrict__ fcb,    // (1) f32
    float*          __restrict__ out)    // (24,512) f32
{
    __shared__ _Float16 hc[4][16 * 128];
    __shared__ _Float16 lin[16 * 256];
    __shared__ float    xin[16];

    const int tid  = threadIdx.x;
    const int wv   = tid >> 5;
    const int ln   = tid & 31;
    const int nn   = ln & 15;
    const int hi   = ln >> 4;
    const int row0 = blockIdx.x * 16;
    const int c0   = wv * 16 + nn;

    for (int c = 0; c < 4; ++c) {
        const _Float16* src = hn + ((size_t)c * 512 + row0) * 128;
        for (int i = tid; i < 16 * 128; i += 256) hc[c][i] = src[i];
    }
    if (tid < 16) xin[tid] = x[((size_t)511 * 512 + row0 + tid) * 64];

    // Per-lane constants: biases for the 4 cells, dec_Wih0 columns.
    float bi[4][3], bh[4][3], w0[2][3];
#pragma unroll
    for (int cell = 0; cell < 4; ++cell) {
        const float* bip = dbih + (size_t)cell * 384;
        const float* bhp = dbhh + (size_t)cell * 384;
        bi[cell][0] = bip[c0]; bi[cell][1] = bip[128 + c0]; bi[cell][2] = bip[256 + c0];
        bh[cell][0] = bhp[c0]; bh[cell][1] = bhp[128 + c0]; bh[cell][2] = bhp[256 + c0];
    }
#pragma unroll
    for (int d = 0; d < 2; ++d) {
        const float* wp = dWih0 + (size_t)d * 384;
        w0[d][0] = wp[c0]; w0[d][1] = wp[128 + c0]; w0[d][2] = wp[256 + c0];
    }
    __syncthreads();

    float nf[2][8];
    for (int step = 0; step < 24; ++step) {
        // ---- layer 0 cells (input = scalar xin per row; gi is VALU) ----
        for (int d = 0; d < 2; ++d) {
            v8f hr = vinit(bh[d][0]), hz = vinit(bh[d][1]), hng = vinit(bh[d][2]);
            mm3(hc[d], 128, dWhh + (size_t)d * 384 * 128, 128, 128,
                c0, nn, hi, hr, hz, hng);
#pragma unroll
            for (int v = 0; v < 8; ++v) {
                const int m = v + 8 * hi;
                float xv = xin[m];
                float r  = sigm(xv * w0[d][0] + bi[d][0] + hr[v]);
                float z  = sigm(xv * w0[d][1] + bi[d][1] + hz[v]);
                float ng = tanh_f(xv * w0[d][2] + bi[d][2] + r * hng[v]);
                float ho = (float)hc[d][m * 128 + c0];
                nf[d][v] = (1.f - z) * ng + z * ho;
            }
        }
        __syncthreads();
#pragma unroll
        for (int d = 0; d < 2; ++d)
#pragma unroll
            for (int v = 0; v < 8; ++v) {
                const int m = v + 8 * hi;
                _Float16 hv = (_Float16)nf[d][v];
                hc[d][m * 128 + c0] = hv;
                lin[m * 256 + d * 128 + c0] = hv;
            }
        __syncthreads();

        // ---- layer 1 cells (input = lin, 16x256) ----
        for (int d = 0; d < 2; ++d) {
            v8f ir = vinit(bi[2 + d][0]), iz = vinit(bi[2 + d][1]), inn = vinit(bi[2 + d][2]);
            mm3(lin, 256, dWih1 + (size_t)d * 384 * 256, 256, 256,
                c0, nn, hi, ir, iz, inn);
            v8f hr = vinit(bh[2 + d][0]), hz = vinit(bh[2 + d][1]), hng = vinit(bh[2 + d][2]);
            mm3(hc[2 + d], 128, dWhh + (size_t)(2 + d) * 384 * 128, 128, 128,
                c0, nn, hi, hr, hz, hng);
#pragma unroll
            for (int v = 0; v < 8; ++v) {
                const int m = v + 8 * hi;
                float r  = sigm(ir[v] + hr[v]);
                float z  = sigm(iz[v] + hz[v]);
                float ng = tanh_f(inn[v] + r * hng[v]);
                float ho = (float)hc[2 + d][m * 128 + c0];
                nf[d][v] = (1.f - z) * ng + z * ho;
            }
        }
        __syncthreads();
#pragma unroll
        for (int d = 0; d < 2; ++d)
#pragma unroll
            for (int v = 0; v < 8; ++v) {
                const int m = v + 8 * hi;
                _Float16 hv = (_Float16)nf[d][v];
                hc[2 + d][m * 128 + c0] = hv;
                lin[m * 256 + d * 128 + c0] = hv;
            }
        __syncthreads();

        // ---- FC: pred = lin @ fc_w^T + fc_b; feeds next step's xin ----
        if (tid < 16) {
            float acc = fcb[0];
            for (int k = 0; k < 256; ++k)
                acc += (float)lin[tid * 256 + k] * fcw[k];
            xin[tid] = acc;
            out[step * 512 + row0 + tid] = acc;
        }
        __syncthreads();
    }
}

// ---------------------------------------------------------------------------
// Host-side launch
// ---------------------------------------------------------------------------
extern "C" void kernel_launch(void* const* d_in, const int* in_sizes, int n_in,
                              void* d_out, int out_size, void* d_ws, size_t ws_size,
                              hipStream_t stream) {
    (void)in_sizes; (void)n_in; (void)out_size; (void)ws_size;

    const float* x    = (const float*)d_in[0];
    const float* eW0  = (const float*)d_in[1];   // (2,384,64)
    const float* eW1  = (const float*)d_in[2];   // (2,384,256)
    const float* eWhh = (const float*)d_in[3];   // (2,2,384,128)
    const float* ebih = (const float*)d_in[4];
    const float* ebhh = (const float*)d_in[5];
    const float* dW0  = (const float*)d_in[6];   // (2,384,1)
    const float* dW1  = (const float*)d_in[7];   // (2,384,256)
    const float* dWhh = (const float*)d_in[8];   // (2,2,384,128)
    const float* dbih = (const float*)d_in[9];
    const float* dbhh = (const float*)d_in[10];
    const float* fcw  = (const float*)d_in[11];
    const float* fcb  = (const float*)d_in[12];
    float* out = (float*)d_out;
    _Float16* ws = (_Float16*)d_ws;

    // workspace layout (in f16 elements)
    const size_t O_EW0  = 0;                         // 49152
    const size_t O_EW1  = O_EW0  + 49152;            // 196608
    const size_t O_EWHH = O_EW1  + 196608;           // 196608
    const size_t O_DW1  = O_EWHH + 196608;           // 196608
    const size_t O_DWHH = O_DW1  + 196608;           // 196608
    const size_t O_OUT0 = O_DWHH + 196608;           // 512*512*256
    const size_t O_HN   = O_OUT0 + (size_t)512 * 512 * 256;  // 4*512*128

    // 1) convert weights to f16
    cvt_kernel<<<(49152 + 255) / 256, 256, 0, stream>>>(eW0,  ws + O_EW0,  49152);
    cvt_kernel<<<(196608 + 255) / 256, 256, 0, stream>>>(eW1,  ws + O_EW1,  196608);
    cvt_kernel<<<(196608 + 255) / 256, 256, 0, stream>>>(eWhh, ws + O_EWHH, 196608);
    cvt_kernel<<<(196608 + 255) / 256, 256, 0, stream>>>(dW1,  ws + O_DW1,  196608);
    cvt_kernel<<<(196608 + 255) / 256, 256, 0, stream>>>(dWhh, ws + O_DWHH, 196608);

    // 2) encoder layers (stream-ordered: layer1 sees full layer0 output)
    // LDS: Whh + Wih + 2*xt + 2*h  (f16)
    const size_t smem0 = (size_t)(384 * 128 + 384 * 64 + 2 * 16 * 64 + 2 * 16 * 128) * 2;   // 159744 B
    enc_layer_kernel<<<64, 256, smem0, stream>>>(
        0, 64, x, ws + O_OUT0, ws + O_EW0, ws + O_EWHH, ebih, ebhh, ws + O_HN);
    const size_t smem1 = (size_t)(384 * 128 + 384 * 256 + 2 * 16 * 256 + 2 * 16 * 128) * 2; // 319488 B
    enc_layer_kernel<<<64, 256, smem1, stream>>>(
        1, 256, x, ws + O_OUT0, ws + O_EW1, ws + O_EWHH, ebih, ebhh, ws + O_HN);

    // 3) decoder (batch-tiled, fully internal 24-step loop)
    dec_kernel<<<32, 256, 0, stream>>>(
        x, ws + O_HN, dW0, ws + O_DW1, ws + O_DWHH, dbih, dbhh, fcw, fcb, out);
}
